// CrossAttention_48713519071925
// MI455X (gfx1250) — matrix operand order repaired and checked by reading, not compiled, not run on hardware
//
#include <hip/hip_runtime.h>

typedef __attribute__((ext_vector_type(16))) _Float16 v16h;
typedef __attribute__((ext_vector_type(8)))  _Float16 v8h;
typedef __attribute__((ext_vector_type(4)))  _Float16 v4h;
typedef __attribute__((ext_vector_type(8)))  float    v8f;
typedef __attribute__((ext_vector_type(4)))  float    v4f;

// Problem constants
#define BATCH 4
#define SQ    2048
#define SK    1024
#define DQC   1024   // QUERY_DIM
#define DKC   768    // CONTEXT_DIM
#define NHEAD 8
#define DH    64
#define INNER 512

#define LDA 40   // GEMM LDS row stride in halves (80B, 16B-aligned)
#define LDP 72   // attention P-tile row stride in halves (144B, 16B-aligned)

__device__ __forceinline__ v16h cat16(v8h lo, v8h hi) {
    return __builtin_shufflevector(lo, hi, 0,1,2,3,4,5,6,7,8,9,10,11,12,13,14,15);
}

__device__ __forceinline__ v8f wmma_f16(v16h a, v16h b, v8f c) {
    return __builtin_amdgcn_wmma_f32_16x16x32_f16(false, a, false, b, (short)0, c,
                                                  false, false);
}

// ---------------------------------------------------------------------------
// Tiled GEMM, C[M,N] = A[M,K] * B[K,N].  Workgroup 128 threads (4 waves).
// Block tile 128x64, wave tile 32x64, K step 32, double-buffered LDS.
//   OUT_MODE 0: C stored f16 row-major
//   OUT_MODE 1: C stored f16 scattered as V^T[b][h][d][j] (M=b*SK+j, N=h*DH+d)
//   OUT_MODE 2: C stored f32 row-major with bias add
//   A_F16   : A is f16 (else f32, converted while staging to LDS)
// M % 128 == 0, N % 64 == 0, K % 32 == 0 for all launches here.
// ---------------------------------------------------------------------------
template<int OUT_MODE, bool A_F16>
__global__ __launch_bounds__(128)
void gemm_wmma(const void* __restrict__ Av, const float* __restrict__ Bg,
               void* __restrict__ Cv, const float* __restrict__ bias,
               int M, int N, int K)
{
    __shared__ _Float16 sA[2 * 128 * LDA];
    __shared__ _Float16 sB[2 * 64 * LDA];

    const int tid  = threadIdx.x;
    const int wave = tid >> 5;
    const int lane = tid & 31;
    const int hfl  = lane >> 4;
    const int ln16 = lane & 15;
    const int m0   = blockIdx.y * 128;
    const int n0   = blockIdx.x * 64;

    auto stageA = [&](int k0, int buf) {
        const int cg    = (tid & 7) * 4;
        const int rbase = tid >> 3;           // 0..15
        _Float16* dst = &sA[buf * (128 * LDA)];
#pragma unroll
        for (int p = 0; p < 8; ++p) {
            const int r = rbase + p * 16;     // 0..127
            if (A_F16) {
                const _Float16* Ah = (const _Float16*)Av;
                *(v4h*)&dst[r * LDA + cg] =
                    *(const v4h*)(Ah + (size_t)(m0 + r) * K + k0 + cg);
            } else {
                const float* Af = (const float*)Av;
                v4f x = *(const v4f*)(Af + (size_t)(m0 + r) * K + k0 + cg);
                dst[r * LDA + cg + 0] = (_Float16)x[0];
                dst[r * LDA + cg + 1] = (_Float16)x[1];
                dst[r * LDA + cg + 2] = (_Float16)x[2];
                dst[r * LDA + cg + 3] = (_Float16)x[3];
            }
        }
    };
    auto stageB = [&](int k0, int buf) {
        const int g  = tid & 15;              // n = 4g..4g+3 (coalesced over n)
        const int cb = tid >> 4;              // 0..7
        _Float16* dst = &sB[buf * (64 * LDA)];
#pragma unroll
        for (int p = 0; p < 4; ++p) {
            const int c = cb + p * 8;         // 0..31
            v4f x = *(const v4f*)(Bg + (size_t)(k0 + c) * N + n0 + 4 * g);
            dst[(4 * g + 0) * LDA + c] = (_Float16)x[0];
            dst[(4 * g + 1) * LDA + c] = (_Float16)x[1];
            dst[(4 * g + 2) * LDA + c] = (_Float16)x[2];
            dst[(4 * g + 3) * LDA + c] = (_Float16)x[3];
        }
    };

    v8f acc[2][4] = {};

    const int nk = K >> 5;
    stageA(0, 0);
    stageB(0, 0);

    for (int kt = 0; kt < nk; ++kt) {
        __syncthreads();
        const int cur = kt & 1;
        if (kt + 1 < nk) {
            stageA((kt + 1) * 32, cur ^ 1);
            stageB((kt + 1) * 32, cur ^ 1);
            if (kt + 2 < nk) {  // L2 prefetch of the tile after next
                const size_t pa = (size_t)(m0 + tid) * K + (size_t)(kt + 2) * 32;
                if (A_F16) __builtin_prefetch((const _Float16*)Av + pa, 0, 1);
                else       __builtin_prefetch((const float*)Av + pa, 0, 1);
                __builtin_prefetch(Bg + (size_t)((kt + 2) * 32 + (tid & 31)) * N + n0, 0, 1);
            }
        }
        // wave tile 32x64: 2 A fragments x 4 B fragments -> 8 WMMA
        v16h afrag[2];
#pragma unroll
        for (int mt = 0; mt < 2; ++mt) {
            const _Float16* ap =
                &sA[cur * (128 * LDA) + (wave * 32 + mt * 16 + ln16) * LDA + hfl * 8];
            afrag[mt] = cat16(*(const v8h*)ap, *(const v8h*)(ap + 16));
        }
#pragma unroll
        for (int nt = 0; nt < 4; ++nt) {
            const _Float16* bp =
                &sB[cur * (64 * LDA) + (nt * 16 + ln16) * LDA + hfl * 8];
            v16h b = cat16(*(const v8h*)bp, *(const v8h*)(bp + 16));
            acc[0][nt] = wmma_f16(afrag[0], b, acc[0][nt]);
            acc[1][nt] = wmma_f16(afrag[1], b, acc[1][nt]);
        }
    }

    // ---- epilogue ----
#pragma unroll
    for (int mt = 0; mt < 2; ++mt)
#pragma unroll
        for (int nt = 0; nt < 4; ++nt)
#pragma unroll
            for (int r = 0; r < 8; ++r) {
                const int m = m0 + wave * 32 + mt * 16 + hfl * 8 + r;
                const int n = n0 + nt * 16 + ln16;
                const float v = acc[mt][nt][r];
                if (OUT_MODE == 0) {
                    ((_Float16*)Cv)[(size_t)m * N + n] = (_Float16)v;
                } else if (OUT_MODE == 1) {
                    const int b = m / SK, j = m % SK;
                    const int h = n / DH, d = n % DH;
                    ((_Float16*)Cv)[((size_t)(b * NHEAD + h) * DH + d) * SK + j] =
                        (_Float16)v;
                } else {
                    ((float*)Cv)[(size_t)m * N + n] = v + bias[n];
                }
            }
}

// ---------------------------------------------------------------------------
// Flash attention: softmax(Q K^T * scale) V per (b,h).
// 128 threads = 4 waves; each wave owns 16 query rows; j-blocks of 64.
// Row sums come from an extra WMMA against an all-ones B matrix (P is already
// in A-layout for the PV matmul) -- only the row max needs lane shuffles.
// ---------------------------------------------------------------------------
__global__ __launch_bounds__(128)
void attn_wmma(const _Float16* __restrict__ Qh, const _Float16* __restrict__ Kh,
               const _Float16* __restrict__ Vt, _Float16* __restrict__ Oh)
{
    __shared__ _Float16 sP[4][16 * LDP];   // per-wave P staging (C->A layout)

    const int tid  = threadIdx.x;
    const int wave = tid >> 5;
    const int lane = tid & 31;
    const int hfl  = lane >> 4;
    const int ln16 = lane & 15;

    const int bh = blockIdx.y;             // 0..31
    const int b  = bh / NHEAD;
    const int h  = bh % NHEAD;
    const int i0 = blockIdx.x * 64 + wave * 16;

    // Preload Q fragments with the 1/sqrt(DH) scale folded in.
    v16h sc;
#pragma unroll
    for (int i = 0; i < 16; ++i) sc[i] = (_Float16)0.125f;
    v16h ones;
#pragma unroll
    for (int i = 0; i < 16; ++i) ones[i] = (_Float16)1.0f;

    const _Float16* qbase = Qh + (size_t)(b * SQ + i0 + ln16) * INNER + h * DH;
    v16h qa[2];
#pragma unroll
    for (int kk = 0; kk < 2; ++kk) {
        const _Float16* qp = qbase + kk * 32 + hfl * 8;
        qa[kk] = cat16(*(const v8h*)qp, *(const v8h*)(qp + 16)) * sc;
    }

    float mrow[8], lrow[8];
#pragma unroll
    for (int r = 0; r < 8; ++r) { mrow[r] = -1e30f; lrow[r] = 0.0f; }
    v8f o[4] = {};

    const _Float16* kbase = Kh + (size_t)(b * SK) * INNER + h * DH;
    const _Float16* vbase = Vt + (size_t)bh * DH * SK;

    for (int j0 = 0; j0 < SK; j0 += 64) {
        // ---- S = (Q*scale) K^T on a 16x64 j-tile : 8 WMMA ----
        v8f s[4] = {};
#pragma unroll
        for (int nt = 0; nt < 4; ++nt) {
            const _Float16* kp = kbase + (size_t)(j0 + nt * 16 + ln16) * INNER;
#pragma unroll
            for (int kk = 0; kk < 2; ++kk) {
                const _Float16* bp = kp + kk * 32 + hfl * 8;
                v16h bf = cat16(*(const v8h*)bp, *(const v8h*)(bp + 16));
                s[nt] = wmma_f16(qa[kk], bf, s[nt]);
            }
        }

        // ---- row max across the 64-j tile (rows live on 16-lane halves) ----
        float tmax[8];
#pragma unroll
        for (int r = 0; r < 8; ++r)
            tmax[r] = fmaxf(fmaxf(s[0][r], s[1][r]), fmaxf(s[2][r], s[3][r]));
#pragma unroll
        for (int off = 1; off < 16; off <<= 1)
#pragma unroll
            for (int r = 0; r < 8; ++r)
                tmax[r] = fmaxf(tmax[r], __shfl_xor(tmax[r], off, 32));

        float alpha[8];
#pragma unroll
        for (int r = 0; r < 8; ++r) {
            const float mn = fmaxf(mrow[r], tmax[r]);
            alpha[r] = __expf(mrow[r] - mn);
            mrow[r]  = mn;
        }

        // ---- P = exp(S - m) staged to wave-private LDS in A-layout ----
        _Float16* pw = &sP[wave][0];
#pragma unroll
        for (int r = 0; r < 8; ++r) {
            const int m = hfl * 8 + r;
#pragma unroll
            for (int nt = 0; nt < 4; ++nt)
                pw[m * LDP + nt * 16 + ln16] =
                    (_Float16)__expf(s[nt][r] - mrow[r]);
        }

        // ---- P A-fragments (shared by row-sum WMMA and PV WMMA) ----
        v16h pa[2];
#pragma unroll
        for (int kk = 0; kk < 2; ++kk) {
            const _Float16* pr = &sP[wave][ln16 * LDP + kk * 32 + hfl * 8];
            pa[kk] = cat16(*(const v8h*)pr, *(const v8h*)(pr + 16));
        }

        // ---- row sums via WMMA against all-ones B : 2 WMMA ----
        v8f rs = {};
        rs = wmma_f16(pa[0], ones, rs);
        rs = wmma_f16(pa[1], ones, rs);
#pragma unroll
        for (int r = 0; r < 8; ++r)
            lrow[r] = lrow[r] * alpha[r] + rs[r];

        // ---- rescale accumulated O, then O += P V : 8 WMMA ----
#pragma unroll
        for (int dt = 0; dt < 4; ++dt)
#pragma unroll
            for (int r = 0; r < 8; ++r)
                o[dt][r] *= alpha[r];
#pragma unroll
        for (int dt = 0; dt < 4; ++dt) {
            const _Float16* vrow = vbase + (size_t)(dt * 16 + ln16) * SK + j0;
#pragma unroll
            for (int kk = 0; kk < 2; ++kk) {
                const _Float16* vp = vrow + kk * 32 + hfl * 8;
                v16h bf = cat16(*(const v8h*)vp, *(const v8h*)(vp + 16));
                o[dt] = wmma_f16(pa[kk], bf, o[dt]);
            }
        }
    }

    // ---- normalize and store ----
#pragma unroll
    for (int r = 0; r < 8; ++r) lrow[r] = 1.0f / lrow[r];
#pragma unroll
    for (int dt = 0; dt < 4; ++dt)
#pragma unroll
        for (int r = 0; r < 8; ++r) {
            const int i = i0 + hfl * 8 + r;
            const int d = dt * 16 + ln16;
            Oh[(size_t)(b * SQ + i) * INNER + h * DH + d] =
                (_Float16)(o[dt][r] * lrow[r]);
        }
}

// ---------------------------------------------------------------------------
extern "C" void kernel_launch(void* const* d_in, const int* in_sizes, int n_in,
                              void* d_out, int out_size, void* d_ws, size_t ws_size,
                              hipStream_t stream)
{
    (void)in_sizes; (void)n_in; (void)out_size; (void)ws_size;

    const float* x   = (const float*)d_in[0];
    const float* ctx = (const float*)d_in[1];
    const float* Wq  = (const float*)d_in[2];
    const float* Wk  = (const float*)d_in[3];
    const float* Wv  = (const float*)d_in[4];
    const float* Wo  = (const float*)d_in[5];
    const float* bo  = (const float*)d_in[6];
    float* out = (float*)d_out;

    char* ws = (char*)d_ws;
    _Float16* Qh = (_Float16*)(ws);                              //  8 MiB: [8192][512]
    _Float16* Kh = (_Float16*)(ws + (size_t)8  * 1024 * 1024);   //  4 MiB: [4096][512]
    _Float16* Vt = (_Float16*)(ws + (size_t)12 * 1024 * 1024);   //  4 MiB: [32*64][1024]
    _Float16* Oh = (_Float16*)(ws + (size_t)16 * 1024 * 1024);   //  8 MiB: [8192][512]

    const dim3 blk(128);

    // Q = x @ Wq            M=8192 N=512 K=1024
    gemm_wmma<0, false><<<dim3(512 / 64, 8192 / 128), blk, 0, stream>>>(
        x, Wq, Qh, nullptr, BATCH * SQ, INNER, DQC);
    // K = ctx @ Wk          M=4096 N=512 K=768
    gemm_wmma<0, false><<<dim3(512 / 64, 4096 / 128), blk, 0, stream>>>(
        ctx, Wk, Kh, nullptr, BATCH * SK, INNER, DKC);
    // V^T = scatter(ctx @ Wv)
    gemm_wmma<1, false><<<dim3(512 / 64, 4096 / 128), blk, 0, stream>>>(
        ctx, Wv, Vt, nullptr, BATCH * SK, INNER, DKC);
    // attention
    attn_wmma<<<dim3(SQ / 64, BATCH * NHEAD), blk, 0, stream>>>(Qh, Kh, Vt, Oh);
    // out = O @ Wo + bo     M=8192 N=1024 K=512 (A f16, f32 out + bias)
    gemm_wmma<2, true><<<dim3(1024 / 64, 8192 / 128), blk, 0, stream>>>(
        Oh, Wo, out, bo, BATCH * SQ, DQC, INNER);
}
